// ssGCN_75256416961205
// MI455X (gfx1250) — compile-verified, matrix operand order checked by reading, not run on hardware
//
#include <hip/hip_runtime.h>
#include <hip/hip_bf16.h>

typedef __attribute__((ext_vector_type(16))) __bf16 v16bf;
typedef __attribute__((ext_vector_type(8)))  __bf16 v8bf;
typedef __attribute__((ext_vector_type(8)))  float  v8f;
typedef __attribute__((ext_vector_type(4)))  float  v4f;

// ---------------------------------------------------------------- degree / norm

__global__ void k_deg_init(float* __restrict__ deg, int n) {
  int i = blockIdx.x * blockDim.x + threadIdx.x;
  if (i < n) deg[i] = 1.0f;                    // self-loop contributes 1
}

__global__ void k_deg_count(const int* __restrict__ dst, float* __restrict__ deg, int e) {
  int i = blockIdx.x * blockDim.x + threadIdx.x;
  if (i < e) atomicAdd(&deg[dst[i]], 1.0f);
}

__global__ void k_deg_rsqrt(float* __restrict__ deg, int n) {
  int i = blockIdx.x * blockDim.x + threadIdx.x;
  if (i < n) deg[i] = rsqrtf(deg[i]);          // deg >= 1 always (self-loop)
}

__global__ void k_norm(const int* __restrict__ src, const int* __restrict__ dst,
                       const float* __restrict__ dinv, float* __restrict__ nrm, int e) {
  int i = blockIdx.x * blockDim.x + threadIdx.x;
  if (i < e) nrm[i] = dinv[src[i]] * dinv[dst[i]];
}

// ---------------------------------------------------------------- bf16 convert / pack

__global__ void k_cvt_bf16(const float* __restrict__ x, __bf16* __restrict__ xb, int count) {
  int i = blockIdx.x * blockDim.x + threadIdx.x;
  if (i < count) xb[i] = (__bf16)x[i];
}

// Pack W (K x N row-major f32) into per-lane WMMA B-operand tiles:
// layout [ntile][ktile][lane 0..31][slot 0..15] contiguous bf16.
// lane: n = base + (lane&15), kbase = (lane>>4)*8
// slot s<8  -> k = kt*32 + kbase + s
// slot s>=8 -> k = kt*32 + kbase + 16 + (s-8)
__global__ void k_pack_w(const float* __restrict__ W, __bf16* __restrict__ Wp, int N, int K) {
  int tid = blockIdx.x * blockDim.x + threadIdx.x;
  int ntiles = N >> 4, ktiles = K >> 5;
  int total = ntiles * ktiles * 32 * 16;
  if (tid >= total) return;
  int s    = tid & 15;
  int lane = (tid >> 4) & 31;
  int tk   = (tid >> 9);                 // ntile*ktiles + kt
  int kt   = tk % ktiles;
  int nt   = tk / ktiles;
  int n    = nt * 16 + (lane & 15);
  int k    = kt * 32 + ((lane >> 4) * 8) + (s < 8 ? s : s + 8);
  Wp[tid] = (__bf16)W[k * N + n];
}

// ---------------------------------------------------------------- WMMA GEMM (bf16 -> f32)
// One wave per 16x16 output tile. A: row-major bf16 [M x K]. Bp: packed tiles.
__global__ void k_gemm_bf16_wmma(const __bf16* __restrict__ A, const __bf16* __restrict__ Bp,
                                 float* __restrict__ C, int M, int N, int K) {
  const int gtid = blockIdx.x * blockDim.x + threadIdx.x;
  const int wave = gtid >> 5;
  const int lane = threadIdx.x & 31;
  const int ntiles = N >> 4;
  const int ktiles = K >> 5;
  const int mt = wave / ntiles;
  const int nt = wave - mt * ntiles;
  if (mt * 16 >= M) return;

  // A-operand addressing per ISA 16-bit A layout (16x32 per ktile)
  const int arow = mt * 16 + (lane & 15);
  const int koff = (lane >> 4) * 8;
  const __bf16* __restrict__ ap = A + (size_t)arow * K;
  const __bf16* __restrict__ bp = Bp + ((size_t)nt * ktiles) * 512;  // 32 lanes * 16 bf16

  v8f acc = {};
  for (int kt = 0; kt < ktiles; ++kt) {
    const int kb = kt * 32 + koff;
    v8bf alo = *(const v8bf*)(ap + kb);
    v8bf ahi = *(const v8bf*)(ap + kb + 16);
    const __bf16* __restrict__ bl = bp + kt * 512 + lane * 16;
    v8bf blo = *(const v8bf*)(bl);
    v8bf bhi = *(const v8bf*)(bl + 8);
    v16bf a, b;
#pragma unroll
    for (int i = 0; i < 8; ++i) { a[i] = alo[i]; a[i + 8] = ahi[i]; b[i] = blo[i]; b[i + 8] = bhi[i]; }
    acc = __builtin_amdgcn_wmma_f32_16x16x32_bf16(false, a, false, b, (short)0, acc,
                                                  false, false);
  }

  // f32 C/D layout: lane -> col = nt*16 + (lane&15); VGPR r -> row = r + (lane>=16 ? 8 : 0)
  const int ncol = nt * 16 + (lane & 15);
  const int rofs = (lane >> 4) * 8;
#pragma unroll
  for (int r = 0; r < 8; ++r) {
    int mr = mt * 16 + r + rofs;
    C[(size_t)mr * N + ncol] = acc[r];
  }
}

// ---------------------------------------------------------------- aggregation

// agg[i][:] = h[i][:] * dinv[i]^2   (self-loop message, avoids 50k atomic edges)
__global__ void k_agg_init(const float* __restrict__ h, const float* __restrict__ dinv,
                           float* __restrict__ agg, int n, int F) {
  int i = blockIdx.x * blockDim.x + threadIdx.x;
  if (i >= n * F) return;
  int node = i / F;
  float d = dinv[node];
  agg[i] = h[i] * d * d;
}

// one thread per (edge, float4 chunk); cshift = log2(F/4)
__global__ void k_scatter(const int* __restrict__ src, const int* __restrict__ dst,
                          const float* __restrict__ nrm, const float* __restrict__ h,
                          float* __restrict__ agg, int e, int F, int cshift) {
  int tid = blockIdx.x * blockDim.x + threadIdx.x;
  int ed = tid >> cshift;
  if (ed >= e) return;
  int c  = tid & ((1 << cshift) - 1);
  int s  = src[ed], d = dst[ed];
  float w = nrm[ed];
  v4f m = *(const v4f*)(h + (size_t)s * F + c * 4);
  float* o = agg + (size_t)d * F + c * 4;
  atomicAdd(o + 0, m[0] * w);
  atomicAdd(o + 1, m[1] * w);
  atomicAdd(o + 2, m[2] * w);
  atomicAdd(o + 3, m[3] * w);
}

// hb[i][j] = bf16(relu(agg[i][j] + bias[j]))
__global__ void k_act_bf16(const float* __restrict__ agg, const float* __restrict__ bias,
                           __bf16* __restrict__ hb, int n, int F) {
  int i = blockIdx.x * blockDim.x + threadIdx.x;
  if (i >= n * F) return;
  int j = i % F;
  hb[i] = (__bf16)fmaxf(agg[i] + bias[j], 0.0f);
}

// ---------------------------------------------------------------- fused FC + log_softmax
// 16 lanes per node (2 nodes per wave32): lane o computes logit o, then width-16
// shuffle reductions for max / sum-exp.
__global__ void k_final(const float* __restrict__ agg2, const float* __restrict__ b2,
                        const float* __restrict__ Wfc, const float* __restrict__ bfc,
                        float* __restrict__ out, int n) {
  int gid = blockIdx.x * blockDim.x + threadIdx.x;
  int node = gid >> 4;
  int o = gid & 15;
  if (node >= n) return;
  const float* __restrict__ a = agg2 + (size_t)node * 64;
  float acc = bfc[o];
#pragma unroll
  for (int j = 0; j < 64; ++j) {
    float v = fmaxf(a[j] + b2[j], 0.0f);
    acc = fmaf(v, Wfc[j * 16 + o], acc);
  }
  float m = acc;
#pragma unroll
  for (int off = 8; off > 0; off >>= 1) m = fmaxf(m, __shfl_xor(m, off, 16));
  float ex = __expf(acc - m);
  float s = ex;
#pragma unroll
  for (int off = 8; off > 0; off >>= 1) s += __shfl_xor(s, off, 16);
  out[(size_t)node * 16 + o] = acc - m - __logf(s);
}

// ---------------------------------------------------------------- launch

static inline int cdiv(long a, long b) { return (int)((a + b - 1) / b); }

extern "C" void kernel_launch(void* const* d_in, const int* in_sizes, int n_in,
                              void* d_out, int out_size, void* d_ws, size_t ws_size,
                              hipStream_t stream) {
  const float* x   = (const float*)d_in[0];
  const int*   ei  = (const int*)d_in[1];
  const float* W1  = (const float*)d_in[2];
  const float* b1  = (const float*)d_in[3];
  const float* W2  = (const float*)d_in[4];
  const float* b2  = (const float*)d_in[5];
  const float* Wfc = (const float*)d_in[6];
  const float* bfc = (const float*)d_in[7];
  float* out = (float*)d_out;

  const int FIN = 128, H1 = 128, H2 = 64;
  const int N = in_sizes[0] / FIN;          // 50000
  const int E = in_sizes[1] / 2;            // 600000
  const int* src = ei;
  const int* dst = ei + E;

  // ---- workspace carve (256B aligned) ----
  char* p = (char*)d_ws;
  size_t off = 0;
  auto carve = [&](size_t bytes) {
    void* r = p + off;
    off += (bytes + 255) & ~(size_t)255;
    return r;
  };
  float*  dinv  = (float*)carve((size_t)N * 4);
  float*  nrm   = (float*)carve((size_t)E * 4);
  __bf16* xb    = (__bf16*)carve((size_t)N * FIN * 2);   // later reused as relu(h1) bf16
  __bf16* w1p   = (__bf16*)carve((size_t)(H1 / 16) * (FIN / 32) * 512 * 2);
  __bf16* w2p   = (__bf16*)carve((size_t)(H2 / 16) * (H1 / 32) * 512 * 2);
  float*  bufA  = (float*)carve((size_t)N * FIN * 4);    // h1, later h2 (N x 64)
  float*  bufB  = (float*)carve((size_t)N * FIN * 4);    // agg1, later agg2 (N x 64)
  (void)ws_size;

  const int T = 256;

  // 1. symmetric normalization coefficients
  k_deg_init<<<cdiv(N, T), T, 0, stream>>>(dinv, N);
  k_deg_count<<<cdiv(E, T), T, 0, stream>>>(dst, dinv, E);
  k_deg_rsqrt<<<cdiv(N, T), T, 0, stream>>>(dinv, N);
  k_norm<<<cdiv(E, T), T, 0, stream>>>(src, dst, dinv, nrm, E);

  // 2. bf16 conversion + weight packing
  k_cvt_bf16<<<cdiv((long)N * FIN, T), T, 0, stream>>>(x, xb, N * FIN);
  k_pack_w<<<cdiv((H1 / 16) * (FIN / 32) * 512, T), T, 0, stream>>>(W1, w1p, H1, FIN);
  k_pack_w<<<cdiv((H2 / 16) * (H1 / 32) * 512, T), T, 0, stream>>>(W2, w2p, H2, H1);

  // 3. layer 1: h1 = xb @ W1   (WMMA bf16 -> f32)
  {
    long waves = (long)(N / 16) * (H1 / 16);
    k_gemm_bf16_wmma<<<cdiv(waves * 32, T), T, 0, stream>>>(xb, w1p, bufA, N, H1, FIN);
  }
  // 4. aggregate layer 1
  k_agg_init<<<cdiv((long)N * H1, T), T, 0, stream>>>(bufA, dinv, bufB, N, H1);
  k_scatter<<<cdiv((long)E * (H1 / 4), T), T, 0, stream>>>(src, dst, nrm, bufA, bufB, E, H1, 5);
  // 5. bias + relu -> bf16 (reuse xb)
  k_act_bf16<<<cdiv((long)N * H1, T), T, 0, stream>>>(bufB, b1, xb, N, H1);

  // 6. layer 2: h2 = relu(h1) @ W2
  {
    long waves = (long)(N / 16) * (H2 / 16);
    k_gemm_bf16_wmma<<<cdiv(waves * 32, T), T, 0, stream>>>(xb, w2p, bufA, N, H2, H1);
  }
  // 7. aggregate layer 2
  k_agg_init<<<cdiv((long)N * H2, T), T, 0, stream>>>(bufA, dinv, bufB, N, H2);
  k_scatter<<<cdiv((long)E * (H2 / 4), T), T, 0, stream>>>(src, dst, nrm, bufA, bufB, E, H2, 4);

  // 8. fused bias+relu + FC(64x16) + log_softmax
  k_final<<<cdiv((long)N * 16, T), T, 0, stream>>>(bufB, b2, Wfc, bfc, out, N);
}
